// SpikeTemporalConvNet2D_71038759076509
// MI455X (gfx1250) — matrix operand order repaired and checked by reading, not compiled, not run on hardware
//
#include <hip/hip_runtime.h>
#include <hip/hip_bf16.h>
#include <cstdint>
#include <cstddef>

// ---------------- constants ----------------
#define ARS 160     // padded activation row stride (positions per bd row)
#define BD_ 256     // B*D = 8*32
#define L_  128
#define LT_ 144     // GEMM column tiling (multiple of 16, >= max Lp=136)

typedef __attribute__((ext_vector_type(16))) __bf16 bf16x16;
typedef __attribute__((ext_vector_type(8)))  __bf16 bf16x8;
typedef __attribute__((ext_vector_type(8)))  float  v8f;

// ---------------- LIF (8 steps, BETA=0.99, THR=1) ----------------
__device__ __forceinline__ float lif_mean8(float x) {
  float mem = 0.f, acc = 0.f;
#pragma unroll
  for (int s = 0; s < 8; ++s) {
    float reset = (mem > 1.0f) ? 1.0f : 0.0f;   // spike(mem - THR)
    mem = 0.99f * mem + x - reset;              // - reset*THR
    acc += (mem > 1.0f) ? 1.0f : 0.0f;
  }
  return acc * 0.125f;
}

// ---------------- WMMA fragment helpers ----------------
// A (16x32 bf16): lanes 0-15: e0..15=K0..15 ; lanes16-31: K16..31
__device__ __forceinline__ bf16x16 load_afrag(const __bf16* wrow, int k0, int half) {
  const __bf16* p = wrow + k0 + half * 16;
  union { bf16x16 v; bf16x8 h[2]; } u;
  u.h[0] = *(const bf16x8*)(p);
  u.h[1] = *(const bf16x8*)(p + 8);
  return u.v;
}

// ---------------- fused conv GEMM (WMMA bf16) + BN-statistics ----------------
// act: [bd][ARS][C] bf16, zero padded.  W: [NTAP][COpad][KS] bf16 (zero padded).
// y:   [COpad][BD_][LT_] f32.  Per-channel sum/sumsq accumulated into s1/s2.
template <int NTAP, int KB>
__global__ __launch_bounds__(32)
void conv_wmma(const __bf16* __restrict__ act, int C, int poff, int dil,
               const __bf16* __restrict__ W, int KS, int COpad,
               const float* __restrict__ bias, int CO_real,
               float* __restrict__ y, int Lp,
               float* __restrict__ s1, float* __restrict__ s2, int do_stats) {
  const int lane = threadIdx.x & 31;
  const int half = lane >> 4;
  const int m    = lane & 15;
  const int co0  = blockIdx.x * 16;
  const int bd   = blockIdx.y;

  bf16x16 afrag[NTAP * KB];
#pragma unroll
  for (int t = 0; t < NTAP; ++t) {
    const __bf16* wrow = W + ((size_t)t * COpad + (co0 + m)) * KS;
#pragma unroll
    for (int kb = 0; kb < KB; ++kb)
      afrag[t * KB + kb] = load_afrag(wrow, kb * 32, half);
  }

  float biasr[8];
#pragma unroll
  for (int r = 0; r < 8; ++r) {
    int co = co0 + r + half * 8;
    biasr[r] = (co < CO_real) ? bias[co] : 0.0f;
  }

  float ssum[8], ssq[8];
#pragma unroll
  for (int r = 0; r < 8; ++r) { ssum[r] = 0.f; ssq[r] = 0.f; }

  const __bf16* actbd = act + (size_t)bd * ARS * C;

  for (int t = 0; t < LT_ / 16; ++t) {
    const int l = t * 16 + m;
    v8f c;
#pragma unroll
    for (int r = 0; r < 8; ++r) c[r] = biasr[r];
#pragma unroll
    for (int tap = 0; tap < NTAP; ++tap) {
      const int p = l + poff - (NTAP - 1 - tap) * dil;   // causal taps
      const __bf16* arow = actbd + (size_t)p * C + half * 16;
#pragma unroll
      for (int kb = 0; kb < KB; ++kb) {
        bf16x16 bfrag = *(const bf16x16*)(arow + kb * 32);
        c = __builtin_amdgcn_wmma_f32_16x16x32_bf16(false, afrag[tap * KB + kb],
                                                    false, bfrag,
                                                    (short)0, c, false, false);
      }
    }
    const bool valid = (l < Lp);
#pragma unroll
    for (int r = 0; r < 8; ++r) {
      int co = co0 + r + half * 8;
      y[((size_t)co * BD_ + bd) * LT_ + l] = c[r];
      if (do_stats && valid) { ssum[r] += c[r]; ssq[r] += c[r] * c[r]; }
    }
  }

  if (do_stats) {
#pragma unroll
    for (int r = 0; r < 8; ++r) {
      float a = ssum[r], b = ssq[r];
      for (int off = 1; off < 16; off <<= 1) {
        a += __shfl_xor(a, off, 32);
        b += __shfl_xor(b, off, 32);
      }
      if (m == 0) {
        int co = co0 + r + half * 8;
        atomicAdd(&s1[co], a);
        atomicAdd(&s2[co], b);
      }
    }
  }
}

// ---------------- fused BatchNorm + LIF ----------------
__global__ __launch_bounds__(256)
void bn_lif(const float* __restrict__ y,
            const float* __restrict__ s1, const float* __restrict__ s2, float invN,
            const float* __restrict__ gamma, const float* __restrict__ beta, int CO,
            __bf16* __restrict__ outp, int C_out, int poff,      // padded bf16 spikes (or null)
            float* __restrict__ outf) {                          // f32 spikes [c][bd][l] (or null)
  int idx = blockIdx.x * blockDim.x + threadIdx.x;
  if (idx >= CO * BD_ * L_) return;
  int l  = idx % L_;
  int bd = (idx / L_) % BD_;
  int c  = idx / (L_ * BD_);
  float mean = s1[c] * invN;
  float var  = s2[c] * invN - mean * mean;
  float sc   = gamma[c] * rsqrtf(var + 1e-5f);
  float v    = (y[((size_t)c * BD_ + bd) * LT_ + l] - mean) * sc + beta[c];
  float o    = lif_mean8(v);
  if (outp) outp[((size_t)bd * ARS + poff + l) * C_out + c] = (__bf16)o;
  if (outf) outf[idx] = o;
}

// ---------------- residual add + LIF (+ final output) ----------------
__global__ __launch_bounds__(256)
void res_lif(const float* __restrict__ s2f,
             const float* __restrict__ resf,                     // f32 residual, stride LT_ (or null)
             const __bf16* __restrict__ resid, int C_in, int poff_in, // identity residual (or null)
             int CO,
             __bf16* __restrict__ outp, int C_out, int poff_out, // next block input (or null)
             float* __restrict__ dout) {
  int idx = blockIdx.x * blockDim.x + threadIdx.x;
  if (idx >= CO * BD_ * L_) return;
  int l  = idx % L_;
  int bd = (idx / L_) % BD_;
  int c  = idx / (L_ * BD_);
  float r;
  if (resf) r = resf[((size_t)c * BD_ + bd) * LT_ + l];
  else      r = (float)resid[((size_t)bd * ARS + poff_in + l) * C_in + c];
  float o = lif_mean8(s2f[idx] + r);
  if (outp) outp[((size_t)bd * ARS + poff_out + l) * C_out + c] = (__bf16)o;
  if (dout && c == 0) {
    dout[bd * L_ + l] = o;
    if (l == L_ - 1) dout[BD_ * L_ + bd] = o;
  }
}

// ---------------- weight prep: weight-norm, tap split, bf16, zero pad ----------------
__global__ void pack_w(const float* __restrict__ v, const float* __restrict__ g,
                       int cout, int cin, __bf16* __restrict__ W, int COpad, int KS) {
  int co = blockIdx.x * blockDim.x + threadIdx.x;
  if (co >= COpad) return;
  float s = 0.f;
  if (co < cout) {
    float nrm = 0.f;
    for (int i = 0; i < cin * 2; ++i) { float t = v[co * cin * 2 + i]; nrm += t * t; }
    s = (nrm > 0.f) ? g[co] * rsqrtf(nrm) : 0.f;
  }
  for (int k = 0; k < KS; ++k) {
    float w0 = 0.f, w1 = 0.f;
    if (co < cout && k < cin) {
      w0 = s * v[(co * cin + k) * 2 + 0];
      w1 = s * v[(co * cin + k) * 2 + 1];
    }
    W[(size_t)co * KS + k]            = (__bf16)w0;   // tap 0 (multiplies x[l-dil])
    W[((size_t)COpad + co) * KS + k]  = (__bf16)w1;   // tap 1 (multiplies x[l])
  }
}

__global__ void pack_dw(const float* __restrict__ dw, int cout, int cin,
                        __bf16* __restrict__ W, int COpad, int KS) {
  int co = blockIdx.x * blockDim.x + threadIdx.x;
  if (co >= COpad) return;
  for (int k = 0; k < KS; ++k)
    W[(size_t)co * KS + k] = (co < cout && k < cin) ? (__bf16)dw[co * cin + k] : (__bf16)0.f;
}

// ---------------- input pack: (B,C,D,L) f32 -> [bd][pos][c] bf16 padded ----------------
__global__ __launch_bounds__(256)
void pack_input(const float* __restrict__ x, __bf16* __restrict__ dst, int Cin, int poff) {
  int idx = blockIdx.x * blockDim.x + threadIdx.x;
  int total = 8 * Cin * 32 * L_;
  if (idx >= total) return;
  int l = idx % L_;
  int d = (idx / L_) & 31;
  int c = (idx / (L_ * 32)) % Cin;
  int b = idx / (L_ * 32 * Cin);
  int bd = b * 32 + d;
  dst[((size_t)bd * ARS + poff + l) * Cin + c] = (__bf16)x[idx];
}

// ---------------- host ----------------
extern "C" void kernel_launch(void* const* d_in, const int* in_sizes, int n_in,
                              void* d_out, int out_size, void* d_ws, size_t ws_size,
                              hipStream_t stream) {
  (void)in_sizes; (void)n_in; (void)out_size; (void)ws_size;
  const float* X = (const float*)d_in[0];
  auto F = [&](int i) { return (const float*)d_in[i]; };

  char* ws = (char*)d_ws;
  size_t off = 0;
  float*  stats = (float*)(ws + off);  off += 1024 * sizeof(float);
  __bf16* W     = (__bf16*)(ws + off); off += 69632 * sizeof(__bf16);
  __bf16* ACT0  = (__bf16*)(ws + off); off += (size_t)BD_ * ARS * 128 * sizeof(__bf16);
  __bf16* ACT1  = (__bf16*)(ws + off); off += (size_t)BD_ * ARS * 64  * sizeof(__bf16);
  __bf16* ACT2  = (__bf16*)(ws + off); off += (size_t)BD_ * ARS * 64  * sizeof(__bf16);
  __bf16* ACT3  = (__bf16*)(ws + off); off += (size_t)BD_ * ARS * 64  * sizeof(__bf16);
  __bf16* S     = (__bf16*)(ws + off); off += (size_t)BD_ * ARS * 64  * sizeof(__bf16);
  const size_t zbytes = off;
  float*  Y     = (float*)(ws + off);  off += (size_t)64 * BD_ * LT_ * sizeof(float);
  float*  S2F   = (float*)(ws + off);  off += (size_t)64 * BD_ * L_  * sizeof(float);
  const size_t SBYTES = (size_t)BD_ * ARS * 64 * sizeof(__bf16);

  hipMemsetAsync(d_ws, 0, zbytes, stream);   // zero stats + all padded activation buffers

  // weights: v/g index pairs follow setup_inputs() dict insertion order
  pack_w <<<1, 64, 0, stream>>>(F(1),  F(2),  64, 128, W + 0,     64, 128);
  pack_w <<<1, 64, 0, stream>>>(F(6),  F(7),  64,  64, W + 16384, 64,  64);
  pack_w <<<1, 64, 0, stream>>>(F(13), F(14), 64,  64, W + 24576, 64,  64);
  pack_w <<<1, 64, 0, stream>>>(F(18), F(19), 64,  64, W + 32768, 64,  64);
  pack_w <<<1, 64, 0, stream>>>(F(23), F(24), 64,  64, W + 40960, 64,  64);
  pack_w <<<1, 64, 0, stream>>>(F(28), F(29), 64,  64, W + 49152, 64,  64);
  pack_w <<<1, 16, 0, stream>>>(F(33), F(34),  1,  64, W + 57344, 16,  64);
  pack_w <<<1, 16, 0, stream>>>(F(38), F(39),  1,   1, W + 59392, 16,  32);
  pack_dw<<<1, 64, 0, stream>>>(F(11), 64, 128, W + 60416, 64, 128);
  pack_dw<<<1, 16, 0, stream>>>(F(43),  1,  64, W + 68608, 16,  64);

  pack_input<<<(8 * 128 * 32 * L_ + 255) / 256, 256, 0, stream>>>(X, ACT0, 128, 1);

  dim3 cg(4, BD_), cg1(1, BD_);

  // ---- Block 0: cin=128, cout=64, dil=1, pad=1, Lp=129 ----
  {
    float invN = 1.f / (256.f * 129.f);
    conv_wmma<2, 4><<<cg, 32, 0, stream>>>(ACT0, 128, 1, 1, W + 0,     128, 64, F(3),  64, Y, 129, stats + 0,   stats + 64,  1);
    bn_lif<<<8192, 256, 0, stream>>>(Y, stats + 0,   stats + 64,  invN, F(4),  F(5),  64, S, 64, 1, nullptr);
    conv_wmma<2, 2><<<cg, 32, 0, stream>>>(S,    64,  1, 1, W + 16384, 64,  64, F(8),  64, Y, 129, stats + 128, stats + 192, 1);
    bn_lif<<<8192, 256, 0, stream>>>(Y, stats + 128, stats + 192, invN, F(9),  F(10), 64, nullptr, 0, 0, S2F);
    conv_wmma<1, 4><<<cg, 32, 0, stream>>>(ACT0, 128, 1, 0, W + 60416, 128, 64, F(12), 64, Y, 129, nullptr, nullptr, 0);
    res_lif<<<8192, 256, 0, stream>>>(S2F, Y, nullptr, 0, 0, 64, ACT1, 64, 2, nullptr);
  }
  hipMemsetAsync(S, 0, SBYTES, stream);
  // ---- Block 1: dil=2, pad=2, Lp=130 ----
  {
    float invN = 1.f / (256.f * 130.f);
    conv_wmma<2, 2><<<cg, 32, 0, stream>>>(ACT1, 64, 2, 2, W + 24576, 64, 64, F(15), 64, Y, 130, stats + 256, stats + 320, 1);
    bn_lif<<<8192, 256, 0, stream>>>(Y, stats + 256, stats + 320, invN, F(16), F(17), 64, S, 64, 2, nullptr);
    conv_wmma<2, 2><<<cg, 32, 0, stream>>>(S,    64, 2, 2, W + 32768, 64, 64, F(20), 64, Y, 130, stats + 384, stats + 448, 1);
    bn_lif<<<8192, 256, 0, stream>>>(Y, stats + 384, stats + 448, invN, F(21), F(22), 64, nullptr, 0, 0, S2F);
    res_lif<<<8192, 256, 0, stream>>>(S2F, nullptr, ACT1, 64, 2, 64, ACT2, 64, 4, nullptr);
  }
  hipMemsetAsync(S, 0, SBYTES, stream);
  // ---- Block 2: dil=4, pad=4, Lp=132 ----
  {
    float invN = 1.f / (256.f * 132.f);
    conv_wmma<2, 2><<<cg, 32, 0, stream>>>(ACT2, 64, 4, 4, W + 40960, 64, 64, F(25), 64, Y, 132, stats + 512, stats + 576, 1);
    bn_lif<<<8192, 256, 0, stream>>>(Y, stats + 512, stats + 576, invN, F(26), F(27), 64, S, 64, 4, nullptr);
    conv_wmma<2, 2><<<cg, 32, 0, stream>>>(S,    64, 4, 4, W + 49152, 64, 64, F(30), 64, Y, 132, stats + 640, stats + 704, 1);
    bn_lif<<<8192, 256, 0, stream>>>(Y, stats + 640, stats + 704, invN, F(31), F(32), 64, nullptr, 0, 0, S2F);
    res_lif<<<8192, 256, 0, stream>>>(S2F, nullptr, ACT2, 64, 4, 64, ACT3, 64, 8, nullptr);
  }
  hipMemsetAsync(S, 0, SBYTES, stream);
  // ---- Block 3: cin=64, cout=1 (padded to 16), dil=8, pad=8, Lp=136 ----
  {
    float invN = 1.f / (256.f * 136.f);
    conv_wmma<2, 2><<<cg1, 32, 0, stream>>>(ACT3, 64, 8, 8, W + 57344, 64, 16, F(35), 1, Y, 136, stats + 768, stats + 832, 1);
    bn_lif<<<128, 256, 0, stream>>>(Y, stats + 768, stats + 832, invN, F(36), F(37), 1, S, 16, 8, nullptr);
    conv_wmma<2, 1><<<cg1, 32, 0, stream>>>(S,    16, 8, 8, W + 59392, 32, 16, F(40), 1, Y, 136, stats + 896, stats + 960, 1);
    bn_lif<<<128, 256, 0, stream>>>(Y, stats + 896, stats + 960, invN, F(41), F(42), 1, nullptr, 0, 0, S2F);
    conv_wmma<1, 2><<<cg1, 32, 0, stream>>>(ACT3, 64, 8, 0, W + 68608, 64, 16, F(44), 1, Y, 136, nullptr, nullptr, 0);
    res_lif<<<128, 256, 0, stream>>>(S2F, Y, nullptr, 0, 0, 1, nullptr, 0, 0, (float*)d_out);
  }
}
